// ConvAttnetion_46454366273544
// MI455X (gfx1250) — matrix operand order repaired and checked by reading, not compiled, not run on hardware
//
#include <hip/hip_runtime.h>

// ---------------------------------------------------------------------------
// ConvAttention for MI455X (gfx1250, wave32, WMMA f16 -> f32)
// ---------------------------------------------------------------------------
#define B_    8
#define DIM_  512
#define IMG_  32
#define HEADS_ 8
#define DH_   64
#define N_    1024
#define INNER_ 512

typedef __attribute__((ext_vector_type(16))) _Float16 v16h;
typedef __attribute__((ext_vector_type(8)))  _Float16 v8h;
typedef __attribute__((ext_vector_type(8)))  float    v8f;

__device__ __forceinline__ v16h cat16(v8h lo, v8h hi) {
  v16h r;
#pragma unroll
  for (int i = 0; i < 8; ++i) { r[i] = lo[i]; r[i + 8] = hi[i]; }
  return r;
}

// A fragment: 16x32 f16 tile, row-major source (ld elements per row).
// lane L: row m = L%16; K run [half*16 .. +15]
__device__ __forceinline__ v16h load_afrag(const _Float16* p, int ld) {
  const int lane = threadIdx.x & 31;
  const _Float16* row = p + (size_t)(lane & 15) * ld + ((lane >> 4) * 16);
  v8h lo = *(const v8h*)(row);
  v8h hi = *(const v8h*)(row + 8);
  return cat16(lo, hi);
}

// B fragment: B is 32x16 (KxN); source is Bt row-major: 16 rows (out col) x 32 (K).
// lane L: out col n = L%16; K run [half*16 .. +15]
__device__ __forceinline__ v16h load_bfrag(const _Float16* bt, int ld) {
  const int lane = threadIdx.x & 31;
  const _Float16* row = bt + (size_t)(lane & 15) * ld + ((lane >> 4) * 16);
  v8h lo = *(const v8h*)(row);
  v8h hi = *(const v8h*)(row + 8);
  return cat16(lo, hi);
}

__device__ __forceinline__ v8f wmma_f16f32(v16h a, v16h b, v8f c) {
  return __builtin_amdgcn_wmma_f32_16x16x32_f16(false, a, false, b, (short)0, c,
                                                false, false);
}

// ---------------------------------------------------------------------------
// Kernel 0: weight conversion to f16 + BN constant folding
// ---------------------------------------------------------------------------
__global__ void __launch_bounds__(256)
k_prep(const float* __restrict__ pw_w, const float* __restrict__ out_w,
       const float* __restrict__ dw_b, const float* __restrict__ bn_g,
       const float* __restrict__ bn_b, const float* __restrict__ bn_m,
       const float* __restrict__ bn_v, _Float16* __restrict__ pw16,
       _Float16* __restrict__ ow16, float* __restrict__ a,
       float* __restrict__ b2) {
  int g = blockIdx.x * 256 + threadIdx.x;
  if (g < 3 * INNER_ * DIM_) pw16[g] = (_Float16)pw_w[g];
  if (g < DIM_ * INNER_)     ow16[g] = (_Float16)out_w[g];
  if (g < 3 * DIM_) {
    float sc = bn_g[g] * rsqrtf(bn_v[g] + 1e-5f);
    a[g]  = sc;
    b2[g] = (dw_b[g] - bn_m[g]) * sc + bn_b[g];
  }
}

// ---------------------------------------------------------------------------
// Kernel 1: depthwise 3x3 conv + folded BN -> yt[i*B+b][n][c] (f16)
// c innermost in both x and yt -> fully coalesced.
// ---------------------------------------------------------------------------
__global__ void __launch_bounds__(256)
k_dwconv(const float* __restrict__ x, const float* __restrict__ dw_w,
         const float* __restrict__ bn_a, const float* __restrict__ bn_b2,
         _Float16* __restrict__ yt) {
  int g  = blockIdx.x * 256 + threadIdx.x;  // 3*B*N*DIM threads
  int c  = g & (DIM_ - 1);
  int n  = (g >> 9) & (N_ - 1);
  int ib = g >> 19;                 // i*B + b
  int i  = ib >> 3, b = ib & 7;
  int h  = n >> 5, w = n & 31;
  const float* wp = dw_w + (size_t)(i * DIM_ + c) * 9;
  float acc = 0.f;
#pragma unroll
  for (int dy = -1; dy <= 1; ++dy) {
    int hh = h + dy;
    if (hh < 0 || hh >= IMG_) continue;
#pragma unroll
    for (int dx = -1; dx <= 1; ++dx) {
      int ww = w + dx;
      if (ww < 0 || ww >= IMG_) continue;
      acc += wp[(dy + 1) * 3 + (dx + 1)] *
             x[((size_t)b * N_ + hh * IMG_ + ww) * DIM_ + c];
    }
  }
  int ic = i * DIM_ + c;
  yt[((size_t)ib * N_ + n) * DIM_ + c] = (_Float16)(acc * bn_a[ic] + bn_b2[ic]);
}

// ---------------------------------------------------------------------------
// Kernel 2: pointwise GEMM  qkv[ib][n][o] = pw_w[i](o,c) @ yt[ib](c,n) + pw_b
// Register-blocked 1x4: one A (weight) fragment feeds 4 B tiles -> 4 WMMAs
// per A load. K = 512 (16 k-steps). global_prefetch for the k-tile 2 ahead.
// ---------------------------------------------------------------------------
__global__ void __launch_bounds__(256)
k_qkv(const _Float16* __restrict__ pw16, const float* __restrict__ pw_b,
      const _Float16* __restrict__ yt, _Float16* __restrict__ qkv) {
  int ib   = blockIdx.y;                       // 0..23
  int wave = threadIdx.x >> 5, lane = threadIdx.x & 31;
  int t    = blockIdx.x * 8 + wave;            // 0..511
  int ot   = t & 31;                           // o tile (32)
  int ng   = t >> 5;                           // group of 4 n tiles (16)
  int i    = ib >> 3;
  const _Float16* A  = pw16 + (size_t)i * INNER_ * DIM_ + (size_t)ot * 16 * DIM_;
  const _Float16* Bt = yt + ((size_t)ib * N_ + ng * 64) * DIM_;
  v8f acc0 = {}, acc1 = {}, acc2 = {}, acc3 = {};
#pragma unroll 2
  for (int kk = 0; kk < DIM_ / 32; ++kk) {
    if (kk + 2 < DIM_ / 32) {
      __builtin_prefetch(A + (kk + 2) * 32, 0, 0);
      __builtin_prefetch(Bt + (kk + 2) * 32, 0, 0);
    }
    v16h af  = load_afrag(A + kk * 32, DIM_);
    v16h bf0 = load_bfrag(Bt + (size_t)(0 * 16) * DIM_ + kk * 32, DIM_);
    v16h bf1 = load_bfrag(Bt + (size_t)(1 * 16) * DIM_ + kk * 32, DIM_);
    v16h bf2 = load_bfrag(Bt + (size_t)(2 * 16) * DIM_ + kk * 32, DIM_);
    v16h bf3 = load_bfrag(Bt + (size_t)(3 * 16) * DIM_ + kk * 32, DIM_);
    acc0 = wmma_f16f32(af, bf0, acc0);
    acc1 = wmma_f16f32(af, bf1, acc1);
    acc2 = wmma_f16f32(af, bf2, acc2);
    acc3 = wmma_f16f32(af, bf3, acc3);
  }
  int half = lane >> 4, col = lane & 15;
  int obase = ot * 16 + half * 8;
  float pb[8];
#pragma unroll
  for (int r = 0; r < 8; ++r) pb[r] = pw_b[i * INNER_ + obase + r];
  v8f* accs[4] = {&acc0, &acc1, &acc2, &acc3};
#pragma unroll
  for (int sub = 0; sub < 4; ++sub) {
    int n = ng * 64 + sub * 16 + col;
    v8h sv;
#pragma unroll
    for (int r = 0; r < 8; ++r) sv[r] = (_Float16)((*accs[sub])[r] + pb[r]);
    *(v8h*)(qkv + ((size_t)ib * N_ + n) * INNER_ + obase) = sv;
  }
}

// ---------------------------------------------------------------------------
// Kernel 3: flash attention. Block = 8 waves; each wave owns 16 q rows of one
// (b,h). Loop over 32-wide K/V tiles staged in LDS (V transposed). Online
// softmax; P routed D-layout -> A-layout via per-wave LDS scratch.
// ---------------------------------------------------------------------------
__global__ void __launch_bounds__(256)
k_attn(const _Float16* __restrict__ qkv, _Float16* __restrict__ attn) {
  __shared__ _Float16 k_sm[32 * 72];       // k[n_k][d], padded rows
  __shared__ _Float16 vt_sm[64 * 40];      // v^T[d][n_k], padded rows
  __shared__ _Float16 p_sm[8 * 16 * 40];   // per-wave P tile (16x32, padded)

  int bh = blockIdx.x;
  int b = bh >> 3, h = bh & 7;
  int wave = threadIdx.x >> 5, lane = threadIdx.x & 31;
  int half = lane >> 4, col = lane & 15;
  int qbase = blockIdx.y * 128 + wave * 16;

  const _Float16* qp = qkv + ((size_t)(0 * B_ + b) * N_) * INNER_ + h * DH_;
  const _Float16* kp = qkv + ((size_t)(1 * B_ + b) * N_) * INNER_ + h * DH_;
  const _Float16* vp = qkv + ((size_t)(2 * B_ + b) * N_) * INNER_ + h * DH_;

  v16h qa0 = load_afrag(qp + (size_t)qbase * INNER_, INNER_);
  v16h qa1 = load_afrag(qp + (size_t)qbase * INNER_ + 32, INNER_);

  float mrow[8], lrow[8];
  v8f o0 = {}, o1 = {}, o2 = {}, o3 = {};
#pragma unroll
  for (int r = 0; r < 8; ++r) { mrow[r] = -1e30f; lrow[r] = 0.f; }
  _Float16* pw = p_sm + wave * 640;

  for (int j = 0; j < N_ / 32; ++j) {
    __syncthreads();
    {  // cooperative stage of K (row-major) and V (transposed) tiles
      int row = threadIdx.x >> 3, ch = threadIdx.x & 7;
      const _Float16* ksrc = kp + (size_t)(j * 32 + row) * INNER_ + ch * 8;
      *(v8h*)&k_sm[row * 72 + ch * 8] = *(const v8h*)ksrc;
      v8h vv = *(const v8h*)(vp + (size_t)(j * 32 + row) * INNER_ + ch * 8);
#pragma unroll
      for (int e = 0; e < 8; ++e) vt_sm[(ch * 8 + e) * 40 + row] = vv[e];
    }
    __syncthreads();

    // S = q @ k^T  (16x32 tile = two 16x16 D frags, K=64 -> 2 WMMAs each)
    v8f s0 = {}, s1 = {};
    {
      const _Float16* kr0 = &k_sm[col * 72];
      const _Float16* kr1 = &k_sm[(16 + col) * 72];
      v16h b00 = cat16(*(const v8h*)(kr0 + half * 16),
                       *(const v8h*)(kr0 + half * 16 + 8));
      v16h b01 = cat16(*(const v8h*)(kr0 + 32 + half * 16),
                       *(const v8h*)(kr0 + 32 + half * 16 + 8));
      v16h b10 = cat16(*(const v8h*)(kr1 + half * 16),
                       *(const v8h*)(kr1 + half * 16 + 8));
      v16h b11 = cat16(*(const v8h*)(kr1 + 32 + half * 16),
                       *(const v8h*)(kr1 + 32 + half * 16 + 8));
      s0 = wmma_f16f32(qa0, b00, s0);
      s0 = wmma_f16f32(qa1, b01, s0);
      s1 = wmma_f16f32(qa0, b10, s1);
      s1 = wmma_f16f32(qa1, b11, s1);
    }

    // online softmax: rows live as element r across 16-lane halves
#pragma unroll
    for (int r = 0; r < 8; ++r) {
      float a0 = s0[r] * 0.125f, a1 = s1[r] * 0.125f;
      float mx = fmaxf(a0, a1);
      mx = fmaxf(mx, __shfl_xor(mx, 1, 32));
      mx = fmaxf(mx, __shfl_xor(mx, 2, 32));
      mx = fmaxf(mx, __shfl_xor(mx, 4, 32));
      mx = fmaxf(mx, __shfl_xor(mx, 8, 32));
      float mn = fmaxf(mrow[r], mx);
      float alpha = __expf(mrow[r] - mn);
      float p0 = __expf(a0 - mn), p1 = __expf(a1 - mn);
      float rs = p0 + p1;
      rs += __shfl_xor(rs, 1, 32);
      rs += __shfl_xor(rs, 2, 32);
      rs += __shfl_xor(rs, 4, 32);
      rs += __shfl_xor(rs, 8, 32);
      lrow[r] = lrow[r] * alpha + rs;
      mrow[r] = mn;
      o0[r] *= alpha; o1[r] *= alpha; o2[r] *= alpha; o3[r] *= alpha;
      pw[(r + 8 * half) * 40 + col]      = (_Float16)p0;
      pw[(r + 8 * half) * 40 + 16 + col] = (_Float16)p1;
    }
    asm volatile("s_wait_dscnt 0x0" ::: "memory");  // P store -> P load (same wave)

    // O += P(16x32) @ V(32x64): one WMMA per 16-wide d tile
    v16h pf = load_afrag(pw, 40);
    o0 = wmma_f16f32(pf, load_bfrag(&vt_sm[0],       40), o0);
    o1 = wmma_f16f32(pf, load_bfrag(&vt_sm[16 * 40], 40), o1);
    o2 = wmma_f16f32(pf, load_bfrag(&vt_sm[32 * 40], 40), o2);
    o3 = wmma_f16f32(pf, load_bfrag(&vt_sm[48 * 40], 40), o3);
  }

  // epilogue: normalize and store attn[b][n][h*64+d] (f16)
#pragma unroll
  for (int r = 0; r < 8; ++r) {
    float inv = 1.f / lrow[r];
    int n = qbase + 8 * half + r;
    size_t base = ((size_t)b * N_ + n) * INNER_ + h * DH_ + col;
    attn[base + 0]  = (_Float16)(o0[r] * inv);
    attn[base + 16] = (_Float16)(o1[r] * inv);
    attn[base + 32] = (_Float16)(o2[r] * inv);
    attn[base + 48] = (_Float16)(o3[r] * inv);
  }
}

// ---------------------------------------------------------------------------
// Kernel 4: final linear  out[b][n][co] = attn[b](n,o) @ out_w(co,o)^T + out_b
// Register-blocked 1x4: one B (weight) fragment feeds 4 A (activation) tiles.
// ---------------------------------------------------------------------------
__global__ void __launch_bounds__(256)
k_out(const _Float16* __restrict__ attn, const _Float16* __restrict__ ow16,
      const float* __restrict__ out_b, float* __restrict__ out) {
  int b = blockIdx.y;
  int wave = threadIdx.x >> 5, lane = threadIdx.x & 31;
  int t = blockIdx.x * 8 + wave;  // 0..511
  int cot = t & 31;               // output-col tile (32)
  int ng  = t >> 5;               // group of 4 n tiles (16)
  const _Float16* A  = attn + ((size_t)b * N_ + ng * 64) * INNER_;
  const _Float16* Bt = ow16 + (size_t)cot * 16 * INNER_;
  v8f acc0 = {}, acc1 = {}, acc2 = {}, acc3 = {};
#pragma unroll 2
  for (int kk = 0; kk < INNER_ / 32; ++kk) {
    if (kk + 2 < INNER_ / 32) {
      __builtin_prefetch(A + (kk + 2) * 32, 0, 0);
      __builtin_prefetch(Bt + (kk + 2) * 32, 0, 0);
    }
    v16h bf  = load_bfrag(Bt + kk * 32, INNER_);
    v16h af0 = load_afrag(A + (size_t)(0 * 16) * INNER_ + kk * 32, INNER_);
    v16h af1 = load_afrag(A + (size_t)(1 * 16) * INNER_ + kk * 32, INNER_);
    v16h af2 = load_afrag(A + (size_t)(2 * 16) * INNER_ + kk * 32, INNER_);
    v16h af3 = load_afrag(A + (size_t)(3 * 16) * INNER_ + kk * 32, INNER_);
    acc0 = wmma_f16f32(af0, bf, acc0);
    acc1 = wmma_f16f32(af1, bf, acc1);
    acc2 = wmma_f16f32(af2, bf, acc2);
    acc3 = wmma_f16f32(af3, bf, acc3);
  }
  int half = lane >> 4, col = lane & 15;
  int co = cot * 16 + col;
  float ob = out_b[co];
  v8f* accs[4] = {&acc0, &acc1, &acc2, &acc3};
#pragma unroll
  for (int sub = 0; sub < 4; ++sub) {
#pragma unroll
    for (int r = 0; r < 8; ++r) {
      int n = ng * 64 + sub * 16 + 8 * half + r;
      out[((size_t)b * N_ + n) * DIM_ + co] = (*accs[sub])[r] + ob;
    }
  }
}

// ---------------------------------------------------------------------------
extern "C" void kernel_launch(void* const* d_in, const int* in_sizes, int n_in,
                              void* d_out, int out_size, void* d_ws,
                              size_t ws_size, hipStream_t stream) {
  (void)in_sizes; (void)n_in; (void)out_size; (void)ws_size;
  const float* x     = (const float*)d_in[0];
  const float* dw_w  = (const float*)d_in[1];
  const float* dw_b  = (const float*)d_in[2];
  const float* bn_g  = (const float*)d_in[3];
  const float* bn_b  = (const float*)d_in[4];
  const float* bn_m  = (const float*)d_in[5];
  const float* bn_v  = (const float*)d_in[6];
  const float* pw_w  = (const float*)d_in[7];
  const float* pw_b  = (const float*)d_in[8];
  const float* out_w = (const float*)d_in[9];
  const float* out_b = (const float*)d_in[10];
  float* out = (float*)d_out;

  char* ws = (char*)d_ws;
  size_t off = 0;
  _Float16* yt   = (_Float16*)(ws + off); off += (size_t)3 * B_ * N_ * DIM_ * 2;
  _Float16* qkv  = (_Float16*)(ws + off); off += (size_t)3 * B_ * N_ * INNER_ * 2;
  _Float16* attn = (_Float16*)(ws + off); off += (size_t)B_ * N_ * INNER_ * 2;
  _Float16* pw16 = (_Float16*)(ws + off); off += (size_t)3 * INNER_ * DIM_ * 2;
  _Float16* ow16 = (_Float16*)(ws + off); off += (size_t)DIM_ * INNER_ * 2;
  float*    bn_a = (float*)(ws + off);    off += (size_t)3 * DIM_ * 4;
  float*    bn_b2= (float*)(ws + off);    off += (size_t)3 * DIM_ * 4;

  k_prep<<<(3 * INNER_ * DIM_ + 255) / 256, 256, 0, stream>>>(
      pw_w, out_w, dw_b, bn_g, bn_b, bn_m, bn_v, pw16, ow16, bn_a, bn_b2);
  k_dwconv<<<(3 * B_ * N_ * DIM_) / 256, 256, 0, stream>>>(x, dw_w, bn_a, bn_b2, yt);
  k_qkv<<<dim3(64, 3 * B_), 256, 0, stream>>>(pw16, pw_b, yt, qkv);
  k_attn<<<dim3(B_ * HEADS_, N_ / 128), 256, 0, stream>>>(qkv, attn);
  k_out<<<dim3(64, B_), 256, 0, stream>>>(attn, ow16, out_b, out);
}
